// FullTimeAtt_61701500174954
// MI455X (gfx1250) — matrix operand order repaired and checked by reading, not compiled, not run on hardware
//
#include <hip/hip_runtime.h>

// ---------------- problem constants ----------------
#define B_   2
#define C_   64
#define FD   256
#define T_   1024
#define RR   8
#define FATT 32      // FD/RR
#define SUPD 8
#define DD   256     // SUPD*FATT
#define HH   8
#define HDIM 32      // DD/HH

typedef __attribute__((ext_vector_type(16))) _Float16 v16h;
typedef __attribute__((ext_vector_type(8)))  _Float16 v8h;
typedef __attribute__((ext_vector_type(8)))  float    v8f;

static __device__ __forceinline__ v8f wmma_f16(const v16h& a, const v16h& b, const v8f& c) {
    return __builtin_amdgcn_wmma_f32_16x16x32_f16(false, a, false, b, (short)0, c, false, false);
}

// ---------------- K0a: adaLN vectors: silu(c) @ W^T + b ----------------
__global__ void k_ada(const float* __restrict__ c,
                      const float* __restrict__ aw, const float* __restrict__ ab,
                      const float* __restrict__ mw, const float* __restrict__ mb,
                      float* __restrict__ adaAtt, float* __restrict__ adaMlp) {
    __shared__ float cs[B_ * 512];
    int tid = threadIdx.x;
    for (int i = tid; i < B_ * 512; i += 256) {
        float v = c[i];
        cs[i] = v / (1.f + __expf(-v));
    }
    __syncthreads();
    for (int idx = tid; idx < B_ * 768 + B_ * 192; idx += 256) {
        if (idx < B_ * 768) {
            int b = idx / 768, o = idx % 768;
            float acc = ab[o];
            const float* w = aw + (size_t)o * 512;
            const float* cc = cs + b * 512;
            for (int k = 0; k < 512; k++) acc += w[k] * cc[k];
            adaAtt[idx] = acc;
        } else {
            int i2 = idx - B_ * 768;
            int b = i2 / 192, o = i2 % 192;
            float acc = mb[o];
            const float* w = mw + (size_t)o * 512;
            const float* cc = cs + b * 512;
            for (int k = 0; k < 512; k++) acc += w[k] * cc[k];
            adaMlp[i2] = acc;
        }
    }
}

// ---------------- K0b: convert matmul weights to f16 ----------------
__global__ void k_cvt(const float* __restrict__ qkv_w, const float* __restrict__ fc1_w,
                      const float* __restrict__ fc2_w,
                      _Float16* __restrict__ wqh, _Float16* __restrict__ w1h,
                      _Float16* __restrict__ w2h) {
    int gid = blockIdx.x * 256 + threadIdx.x;
    const int NQ = 768 * 256, N1 = 256 * 64, N2 = 64 * 128;
    if (gid < NQ)                 wqh[gid]            = (_Float16)qkv_w[gid];
    else if (gid < NQ + N1)       w1h[gid - NQ]       = (_Float16)fc1_w[gid - NQ];
    else if (gid < NQ + N1 + N2)  w2h[gid - NQ - N1]  = (_Float16)fc2_w[gid - NQ - N1];
}

// ---------------- K1: LayerNorm stats over C at each (b,f,t) ----------------
__global__ void k_lnstats(const float* __restrict__ x,
                          float* __restrict__ mean1, float* __restrict__ rstd1) {
    int blk = blockIdx.x;                 // b*1024 + f*4 + tc
    int tc = blk & 3, f = (blk >> 2) & 255, b = blk >> 10;
    int t = tc * 256 + threadIdx.x;
    const float* xp = x + (((size_t)b * C_) * FD + f) * T_ + t;
    float s = 0.f, ss = 0.f;
#pragma unroll 8
    for (int c = 0; c < C_; c++) {
        float v = xp[(size_t)c * FD * T_];
        s += v; ss += v * v;
    }
    float m = s * (1.f / C_);
    float var = ss * (1.f / C_) - m * m;
    size_t o = ((size_t)b * FD + f) * T_ + t;
    mean1[o] = m;
    rstd1[o] = rsqrtf(var + 1e-6f);
}

// ---------------- K2: LN + pixel-unshuffle + sup 1x1 + GLU -> xfr[B,T,D] ----------------
__global__ void k_sup(const float* __restrict__ x, const float* __restrict__ mean1,
                      const float* __restrict__ rstd1, const float* __restrict__ sup_w,
                      const float* __restrict__ sup_b, float* __restrict__ xfr) {
    __shared__ float sw[16 * 512];
    int tid = threadIdx.x;
    for (int i = tid; i < 16 * 512; i += 256) sw[i] = sup_w[i];
    __syncthreads();
    int blk = blockIdx.x;                 // b*128 + fatt*4 + tc
    int tc = blk & 3, fatt = (blk >> 2) & 31, b = blk >> 7;
    int t = tc * 256 + tid;
    float mn[8], rs[8];
#pragma unroll
    for (int r = 0; r < 8; r++) {
        size_t o = ((size_t)b * FD + (fatt * 8 + r)) * T_ + t;
        mn[r] = mean1[o]; rs[r] = rstd1[o];
    }
    float acc[16];
#pragma unroll
    for (int o = 0; o < 16; o++) acc[o] = sup_b[o];
    for (int c = 0; c < C_; c++) {
#pragma unroll
        for (int r = 0; r < 8; r++) {
            int f = fatt * 8 + r;
            float v = x[(((size_t)b * C_ + c) * FD + f) * T_ + t];
            float ln = (v - mn[r]) * rs[r];
            int cc = c * 8 + r;
#pragma unroll
            for (int o = 0; o < 16; o++) acc[o] += sw[o * 512 + cc] * ln;
        }
    }
#pragma unroll
    for (int s2 = 0; s2 < 8; s2++) {
        float g = acc[8 + s2];
        float val = acc[s2] * (g / (1.f + __expf(-g)));
        xfr[((size_t)b * T_ + t) * DD + (s2 * FATT + fatt)] = val;
    }
}

// ---------------- K3: LN over D + modulate -> f16 xh[B,T,D] ----------------
__global__ void k_mod(const float* __restrict__ xfr, const float* __restrict__ adaAtt,
                      _Float16* __restrict__ xh) {
    __shared__ float red[256];
    int blk = blockIdx.x;
    int b = blk / T_, t = blk % T_;
    int d = threadIdx.x;
    float v = xfr[((size_t)b * T_ + t) * DD + d];
    red[d] = v;
    for (int s = 128; s > 0; s >>= 1) { __syncthreads(); if (d < s) red[d] += red[d + s]; }
    __syncthreads();
    float mean = red[0] * (1.f / DD);
    __syncthreads();
    float dv = v - mean;
    red[d] = dv * dv;
    for (int s = 128; s > 0; s >>= 1) { __syncthreads(); if (d < s) red[d] += red[d + s]; }
    __syncthreads();
    float rstd = rsqrtf(red[0] * (1.f / DD) + 1e-6f);
    float sh = adaAtt[b * 768 + d];
    float sc = adaAtt[b * 768 + 256 + d];
    xh[((size_t)b * T_ + t) * DD + d] = (_Float16)(dv * rstd * (1.f + sc) + sh);
}

// ---------------- K4: QKV GEMM with WMMA: qkv[B,T,768] = xh @ Wqkv^T + b ----------------
__global__ void k_qkv(const _Float16* __restrict__ xh, const _Float16* __restrict__ wqh,
                      const float* __restrict__ qkv_b, float* __restrict__ qkvb) {
    int tid = threadIdx.x, lane = tid & 31, w = tid >> 5;
    int widx = blockIdx.x * 8 + w;          // B * 64 mtiles * 48 ntiles
    int b = widx / (64 * 48);
    int r1 = widx % (64 * 48);
    int mt = r1 / 48, nt = r1 % 48;
    int lm = lane & 15, kh2 = lane >> 4;
    int m = mt * 16 + lm;
    int n = nt * 16 + lm;
    v8f cacc;
    float bias = qkv_b[n];
#pragma unroll
    for (int e = 0; e < 8; e++) cacc[e] = bias;
    const _Float16* arow = xh + ((size_t)b * T_ + m) * DD;
#pragma unroll
    for (int k0 = 0; k0 < DD; k0 += 32) {
        v8h lo = *(const v8h*)(arow + k0 + 16 * kh2);
        v8h hi = *(const v8h*)(arow + k0 + 8 + 16 * kh2);
        v16h a;
#pragma unroll
        for (int e = 0; e < 8; e++) { a[e] = lo[e]; a[8 + e] = hi[e]; }
        v16h bb = *(const v16h*)(wqh + (size_t)n * DD + k0 + 16 * kh2);
        cacc = wmma_f16(a, bb, cacc);
    }
#pragma unroll
    for (int e = 0; e < 8; e++) {
        int trow = mt * 16 + kh2 * 8 + e;
        qkvb[((size_t)b * T_ + trow) * 768 + n] = cacc[e];
    }
}

// ---------------- K5: RoPE + repack q,k [B,H,T,hd] f16 and v transposed [B,H,hd,T] ----------------
__global__ void k_rope(const float* __restrict__ qkvb, _Float16* __restrict__ qh,
                       _Float16* __restrict__ kh, _Float16* __restrict__ vt) {
    int gid = blockIdx.x * 256 + threadIdx.x;   // ((b*T+t)*H+h)*16+i
    int i = gid & 15, h = (gid >> 4) & 7, t = (gid >> 7) & 1023, b = gid >> 17;
    size_t row = ((size_t)b * T_ + t) * 768;
    int dbase = h * HDIM + 2 * i;
    float qa = qkvb[row + dbase],        qb = qkvb[row + dbase + 1];
    float ka = qkvb[row + 256 + dbase],  kb = qkvb[row + 256 + dbase + 1];
    float va = qkvb[row + 512 + dbase],  vb = qkvb[row + 512 + dbase + 1];
    float inv = __powf(10000.f, -((2.f * i) / (float)HDIM));
    float sn, cs;
    __sincosf((float)t * inv, &sn, &cs);
    size_t qo = (((size_t)b * HH + h) * T_ + t) * HDIM + 2 * i;
    qh[qo]     = (_Float16)(qa * cs - qb * sn);
    qh[qo + 1] = (_Float16)(qa * sn + qb * cs);
    kh[qo]     = (_Float16)(ka * cs - kb * sn);
    kh[qo + 1] = (_Float16)(ka * sn + kb * cs);
    size_t vo = (((size_t)b * HH + h) * HDIM + 2 * i) * T_ + t;
    vt[vo]      = (_Float16)va;
    vt[vo + T_] = (_Float16)vb;
}

// ---------------- K6: flash attention with WMMA (per-wave 16 q rows, 32-key chunks) ----------------
__global__ void k_attn(const _Float16* __restrict__ qh, const _Float16* __restrict__ kh,
                       const _Float16* __restrict__ vt, float* __restrict__ obuf) {
    __shared__ _Float16 plds[8][16][40];     // bank-safe, 16B-aligned rows
    int tid = threadIdx.x, lane = tid & 31, w = tid >> 5;
    int blk = blockIdx.x;                    // B*H*8
    int b = blk / (HH * 8);
    int r1 = blk % (HH * 8);
    int h = r1 / 8, qt = r1 % 8;
    int m0 = qt * 128 + w * 16;
    size_t bh = (size_t)b * HH + h;
    const _Float16* qp = qh + bh * T_ * HDIM;
    const _Float16* kp = kh + bh * T_ * HDIM;
    const _Float16* vp = vt + bh * HDIM * T_;
    int lm = lane & 15, kh2 = lane >> 4;

    v16h aq;
    {
        v8h lo = *(const v8h*)(qp + (size_t)(m0 + lm) * HDIM + 16 * kh2);
        v8h hi = *(const v8h*)(qp + (size_t)(m0 + lm) * HDIM + 8 + 16 * kh2);
#pragma unroll
        for (int e = 0; e < 8; e++) { aq[e] = lo[e]; aq[8 + e] = hi[e]; }
    }
    v8f o0, o1;
    float mrun[8], lrun[8];
#pragma unroll
    for (int j = 0; j < 8; j++) { o0[j] = 0.f; o1[j] = 0.f; mrun[j] = -1e30f; lrun[j] = 0.f; }
    const float sc = 0.17677669529663687f;   // 1/sqrt(32)

    for (int kt = 0; kt < T_; kt += 32) {
        v8f z;
#pragma unroll
        for (int e = 0; e < 8; e++) z[e] = 0.f;
        v16h bk0 = *(const v16h*)(kp + (size_t)(kt + lm) * HDIM + 16 * kh2);
        v16h bk1 = *(const v16h*)(kp + (size_t)(kt + 16 + lm) * HDIM + 16 * kh2);
        v8f s0 = wmma_f16(aq, bk0, z);
        v8f s1 = wmma_f16(aq, bk1, z);
#pragma unroll
        for (int j = 0; j < 8; j++) {
            float a0 = s0[j] * sc, a1 = s1[j] * sc;
            float v = fmaxf(a0, a1);
#pragma unroll
            for (int off = 1; off < 16; off <<= 1) v = fmaxf(v, __shfl_xor(v, off, 32));
            float nm = fmaxf(mrun[j], v);
            float corr = __expf(mrun[j] - nm);
            float p0 = __expf(a0 - nm), p1 = __expf(a1 - nm);
            float rs = p0 + p1;
#pragma unroll
            for (int off = 1; off < 16; off <<= 1) rs += __shfl_xor(rs, off, 32);
            lrun[j] = lrun[j] * corr + rs;
            mrun[j] = nm;
            o0[j] *= corr; o1[j] *= corr;
            int row = kh2 * 8 + j;
            plds[w][row][lm]      = (_Float16)p0;
            plds[w][row][16 + lm] = (_Float16)p1;
        }
        v16h ap;
#pragma unroll
        for (int e = 0; e < 8; e++) {
            ap[e]     = plds[w][lm][16 * kh2 + e];
            ap[8 + e] = plds[w][lm][8 + 16 * kh2 + e];
        }
        v16h bv0 = *(const v16h*)(vp + (size_t)lm * T_ + kt + 16 * kh2);
        v16h bv1 = *(const v16h*)(vp + (size_t)(16 + lm) * T_ + kt + 16 * kh2);
        o0 = wmma_f16(ap, bv0, o0);
        o1 = wmma_f16(ap, bv1, o1);
    }
#pragma unroll
    for (int j = 0; j < 8; j++) {
        float inv = 1.f / lrun[j];
        int trow = m0 + kh2 * 8 + j;
        size_t base = ((size_t)b * T_ + trow) * DD + h * HDIM;
        obuf[base + lm]      = o0[j] * inv;
        obuf[base + 16 + lm] = o1[j] * inv;
    }
}

// ---------------- K7: out-proj + gate + up-LN -> xg[B,D,T] ----------------
__global__ void k_oproj(const float* __restrict__ obuf, const float* __restrict__ out_w,
                        const float* __restrict__ out_b, const float* __restrict__ adaAtt,
                        float* __restrict__ xg) {
    __shared__ float row[DD];
    __shared__ float red[256];
    int blk = blockIdx.x;
    int b = blk / T_, t = blk % T_;
    int d = threadIdx.x;
    row[d] = obuf[((size_t)b * T_ + t) * DD + d];
    __syncthreads();
    float acc = out_b[d];
    const float* wr = out_w + (size_t)d * DD;
#pragma unroll 8
    for (int k = 0; k < DD; k++) acc += wr[k] * row[k];
    float g = adaAtt[b * 768 + 512 + d] * acc;
    red[d] = g;
    for (int s = 128; s > 0; s >>= 1) { __syncthreads(); if (d < s) red[d] += red[d + s]; }
    __syncthreads();
    float mean = red[0] * (1.f / DD);
    __syncthreads();
    float dv = g - mean;
    red[d] = dv * dv;
    for (int s = 128; s > 0; s >>= 1) { __syncthreads(); if (d < s) red[d] += red[d + s]; }
    __syncthreads();
    float rstd = rsqrtf(red[0] * (1.f / DD) + 1e-6f);
    xg[((size_t)b * DD + d) * T_ + t] = dv * rstd;
}

// ---------------- K8: recover conv(3,1) + pixel-shuffle back + residual -> d_out ----------------
__global__ void k_recov(const float* __restrict__ x, const float* __restrict__ xg,
                        const float* __restrict__ recov_w, const float* __restrict__ recov_b,
                        float* __restrict__ out) {
    int blk = blockIdx.x;                 // ((b*64+co)*256+f)*4+tc
    int tc = blk & 3, f = (blk >> 2) & 255, co = (blk >> 10) & 63, b = blk >> 16;
    int t = tc * 256 + threadIdx.x;
    int r = f & 7, fatt = f >> 3;
    int ch = co * 8 + r;                  // conv output channel
    float acc = recov_b[ch];
#pragma unroll
    for (int kf = 0; kf < 3; kf++) {
        int fi = fatt - 1 + kf;
        if (fi < 0 || fi >= FATT) continue;
#pragma unroll
        for (int s2 = 0; s2 < 8; s2++) {
            acc += recov_w[((size_t)ch * 8 + s2) * 3 + kf] *
                   xg[(((size_t)b * SUPD + s2) * FATT + fi) * T_ + t];
        }
    }
    size_t idx = (((size_t)b * C_ + co) * FD + f) * T_ + t;
    out[idx] = x[idx] + acc;
}

// ---------------- K9: fused MLP (LN + modulate + fc1 WMMA + GLU + fc2 WMMA + residual), in place ----------------
__global__ void k_mlp(float* __restrict__ out, const _Float16* __restrict__ w1h,
                      const _Float16* __restrict__ w2h, const float* __restrict__ fc1_b,
                      const float* __restrict__ fc2_b, const float* __restrict__ adaMlp) {
    __shared__ _Float16 ahs[128 * 72];    // modulated input tile, f16, padded rows
    __shared__ _Float16 hs[128 * 136];    // GLU hidden tile, f16, padded rows
    int tid = threadIdx.x, lane = tid & 31, w = tid >> 5;
    int blk = blockIdx.x;                 // B * 2048 tiles of 128 positions
    int b = blk >> 11, tile = blk & 2047;
    size_t P = (size_t)FD * T_;
    size_t p0 = (size_t)tile * 128;
    const float* adm = adaMlp + b * 192;

    if (tid < 128) {
        size_t base = (size_t)b * C_ * P + p0 + tid;
        float s = 0.f, ss = 0.f;
#pragma unroll 8
        for (int c = 0; c < C_; c++) {
            float v = out[base + (size_t)c * P];
            s += v; ss += v * v;
        }
        float mean = s * (1.f / C_);
        float var = ss * (1.f / C_) - mean * mean;
        float rstd = rsqrtf(var + 1e-6f);
#pragma unroll 8
        for (int c = 0; c < C_; c++) {
            float v = out[base + (size_t)c * P];
            float a = (v - mean) * rstd * (1.f + adm[64 + c]) + adm[c];
            ahs[tid * 72 + c] = (_Float16)a;
        }
    }
    __syncthreads();

    int lm = lane & 15, kh2 = lane >> 4;
    int m = w * 16 + lm;
    v16h a0, a1;
#pragma unroll
    for (int e = 0; e < 8; e++) {
        a0[e]     = ahs[m * 72 + 16 * kh2 + e];
        a0[8 + e] = ahs[m * 72 + 8 + 16 * kh2 + e];
        a1[e]     = ahs[m * 72 + 32 + 16 * kh2 + e];
        a1[8 + e] = ahs[m * 72 + 40 + 16 * kh2 + e];
    }
    // fc1 (64 -> 256) + GLU (-> 128), hidden tile staged in LDS
#pragma unroll
    for (int j = 0; j < 8; j++) {
        v8f ca, cb;
        float ba = fc1_b[j * 16 + lm], bb = fc1_b[128 + j * 16 + lm];
#pragma unroll
        for (int e = 0; e < 8; e++) { ca[e] = ba; cb[e] = bb; }
        int na = j * 16 + lm, nb = 128 + j * 16 + lm;
        v16h b0 = *(const v16h*)(w1h + (size_t)na * 64 + 16 * kh2);
        v16h b1 = *(const v16h*)(w1h + (size_t)na * 64 + 32 + 16 * kh2);
        ca = wmma_f16(a0, b0, ca);
        ca = wmma_f16(a1, b1, ca);
        v16h b2 = *(const v16h*)(w1h + (size_t)nb * 64 + 16 * kh2);
        v16h b3 = *(const v16h*)(w1h + (size_t)nb * 64 + 32 + 16 * kh2);
        cb = wmma_f16(a0, b2, cb);
        cb = wmma_f16(a1, b3, cb);
#pragma unroll
        for (int e = 0; e < 8; e++) {
            float g = cb[e];
            float hv = ca[e] * (g / (1.f + __expf(-g)));
            int row = w * 16 + kh2 * 8 + e;
            hs[row * 136 + j * 16 + lm] = (_Float16)hv;
        }
    }
    // fc2 (128 -> 64); wave reads only rows it wrote -> no block barrier needed
    v16h h0, h1, h2, h3;
#pragma unroll
    for (int e = 0; e < 8; e++) {
        h0[e] = hs[m * 136 + 16 * kh2 + e];       h0[8 + e] = hs[m * 136 + 8 + 16 * kh2 + e];
        h1[e] = hs[m * 136 + 32 + 16 * kh2 + e];  h1[8 + e] = hs[m * 136 + 40 + 16 * kh2 + e];
        h2[e] = hs[m * 136 + 64 + 16 * kh2 + e];  h2[8 + e] = hs[m * 136 + 72 + 16 * kh2 + e];
        h3[e] = hs[m * 136 + 96 + 16 * kh2 + e];  h3[8 + e] = hs[m * 136 + 104 + 16 * kh2 + e];
    }
#pragma unroll
    for (int nt = 0; nt < 4; nt++) {
        v8f c2;
        float bz = fc2_b[nt * 16 + lm];
#pragma unroll
        for (int e = 0; e < 8; e++) c2[e] = bz;
        int n = nt * 16 + lm;
        v16h bb0 = *(const v16h*)(w2h + (size_t)n * 128 + 16 * kh2);
        v16h bb1 = *(const v16h*)(w2h + (size_t)n * 128 + 32 + 16 * kh2);
        v16h bb2 = *(const v16h*)(w2h + (size_t)n * 128 + 64 + 16 * kh2);
        v16h bb3 = *(const v16h*)(w2h + (size_t)n * 128 + 96 + 16 * kh2);
        c2 = wmma_f16(h0, bb0, c2);
        c2 = wmma_f16(h1, bb1, c2);
        c2 = wmma_f16(h2, bb2, c2);
        c2 = wmma_f16(h3, bb3, c2);
#pragma unroll
        for (int e = 0; e < 8; e++) {
            int ch = nt * 16 + lm;
            size_t pos = p0 + w * 16 + kh2 * 8 + e;
            size_t idx = ((size_t)b * C_ + ch) * P + pos;
            out[idx] = out[idx] + adm[128 + ch] * c2[e];
        }
    }
}

// ---------------- host launcher ----------------
extern "C" void kernel_launch(void* const* d_in, const int* in_sizes, int n_in,
                              void* d_out, int out_size, void* d_ws, size_t ws_size,
                              hipStream_t stream) {
    const float* x        = (const float*)d_in[0];
    const float* c        = (const float*)d_in[1];
    const float* sup_w    = (const float*)d_in[2];
    const float* sup_b    = (const float*)d_in[3];
    const float* qkv_w    = (const float*)d_in[4];
    const float* qkv_b    = (const float*)d_in[5];
    const float* out_w    = (const float*)d_in[6];
    const float* out_b    = (const float*)d_in[7];
    const float* recov_w  = (const float*)d_in[8];
    const float* recov_b  = (const float*)d_in[9];
    const float* ada_att_w = (const float*)d_in[10];
    const float* ada_att_b = (const float*)d_in[11];
    const float* ada_mlp_w = (const float*)d_in[12];
    const float* ada_mlp_b = (const float*)d_in[13];
    const float* fc1_w    = (const float*)d_in[14];
    const float* fc1_b    = (const float*)d_in[15];
    const float* fc2_w    = (const float*)d_in[16];
    const float* fc2_b    = (const float*)d_in[17];
    float* outp = (float*)d_out;

    char* ws = (char*)d_ws;
    size_t off = 0;
    auto alloc = [&](size_t bytes) -> void* {
        void* p = ws + off;
        off = (off + bytes + 255) & ~(size_t)255;
        return p;
    };
    const size_t NBFT = (size_t)B_ * FD * T_;     // 524288
    float*    mean1  = (float*)alloc(NBFT * 4);
    float*    rstd1  = (float*)alloc(NBFT * 4);
    float*    adaAtt = (float*)alloc(B_ * 768 * 4);
    float*    adaMlp = (float*)alloc(B_ * 192 * 4);
    float*    xfr    = (float*)alloc((size_t)B_ * T_ * DD * 4);
    _Float16* xh     = (_Float16*)alloc((size_t)B_ * T_ * DD * 2);
    _Float16* wqh    = (_Float16*)alloc((size_t)768 * 256 * 2);
    _Float16* w1h    = (_Float16*)alloc((size_t)256 * 64 * 2);
    _Float16* w2h    = (_Float16*)alloc((size_t)64 * 128 * 2);
    float*    qkvb   = (float*)alloc((size_t)B_ * T_ * 768 * 4);
    _Float16* qh     = (_Float16*)alloc((size_t)B_ * HH * T_ * HDIM * 2);
    _Float16* kh     = (_Float16*)alloc((size_t)B_ * HH * T_ * HDIM * 2);
    _Float16* vt     = (_Float16*)alloc((size_t)B_ * HH * HDIM * T_ * 2);
    float*    obuf   = (float*)alloc((size_t)B_ * T_ * DD * 4);
    float*    xg     = (float*)alloc((size_t)B_ * DD * T_ * 4);
    (void)ws_size; (void)in_sizes; (void)n_in; (void)out_size;

    k_ada<<<1, 256, 0, stream>>>(c, ada_att_w, ada_att_b, ada_mlp_w, ada_mlp_b, adaAtt, adaMlp);
    k_cvt<<<(768 * 256 + 256 * 64 + 64 * 128 + 255) / 256, 256, 0, stream>>>(
        qkv_w, fc1_w, fc2_w, wqh, w1h, w2h);
    k_lnstats<<<B_ * FD * 4, 256, 0, stream>>>(x, mean1, rstd1);
    k_sup<<<B_ * FATT * 4, 256, 0, stream>>>(x, mean1, rstd1, sup_w, sup_b, xfr);
    k_mod<<<B_ * T_, 256, 0, stream>>>(xfr, adaAtt, xh);
    k_qkv<<<(B_ * 64 * 48) / 8, 256, 0, stream>>>(xh, wqh, qkv_b, qkvb);
    k_rope<<<(B_ * T_ * HH * 16) / 256, 256, 0, stream>>>(qkvb, qh, kh, vt);
    k_attn<<<B_ * HH * 8, 256, 0, stream>>>(qh, kh, vt, obuf);
    k_oproj<<<B_ * T_, 256, 0, stream>>>(obuf, out_w, out_b, adaAtt, xg);
    k_recov<<<B_ * C_ * FD * 4, 256, 0, stream>>>(x, xg, recov_w, recov_b, outp);
    k_mlp<<<B_ * 2048, 256, 0, stream>>>(outp, w1h, w2h, fc1_b, fc2_b, adaMlp);
}
